// _DCN_Module_5050881540436
// MI455X (gfx1250) — matrix operand order repaired and checked by reading, not compiled, not run on hardware
//
#include <hip/hip_runtime.h>
#include <hip/hip_bf16.h>

// ---------------------------------------------------------------------------
// DCN step for MI455X (gfx1250, wave32, WMMA):
//   labels  = argmin_k ||x - c_k||^2        (bf16 WMMA GEMM + fused argmin)
//   loss    = sum ||x - c_label||^2 / N     (exact fp32, segment pass)
//   centers = (c0*m0 + seg_sum) / (c0 + n)  (closed-form running mean)
// ---------------------------------------------------------------------------

#define NN 131072
#define KK 2048
#define DD 256

typedef __attribute__((ext_vector_type(16))) __bf16 v16bf;
typedef __attribute__((ext_vector_type(8)))  float  v8f;

// ---------------------------------------------------------------------------
// Kernel 0: zero the atomic-accumulation scratch (seg_sum, n_assigned, loss)
// ---------------------------------------------------------------------------
__global__ __launch_bounds__(256) void dcn_zero_kernel(float* p, int n) {
    int i = blockIdx.x * blockDim.x + threadIdx.x;
    int stride = gridDim.x * blockDim.x;
    for (; i < n; i += stride) p[i] = 0.0f;
}

// ---------------------------------------------------------------------------
// Kernel 1: centers fp32 -> bf16, and c2[k] = ||c_k||^2 (fp32)
// One block per center row; 256 threads = 256 dims.
// ---------------------------------------------------------------------------
__global__ __launch_bounds__(256) void dcn_prep_centers_kernel(
    const float* __restrict__ C, __bf16* __restrict__ Cb, float* __restrict__ c2) {
    const int k = blockIdx.x;
    const int d = threadIdx.x;
    float v = C[(size_t)k * DD + d];
    Cb[(size_t)k * DD + d] = (__bf16)v;
    float s = v * v;
#pragma unroll
    for (int m = 1; m < 32; m <<= 1) s += __shfl_xor(s, m, 32);
    __shared__ float red[8];
    if ((threadIdx.x & 31) == 0) red[threadIdx.x >> 5] = s;
    __syncthreads();
    if (threadIdx.x == 0) {
        float t = 0.f;
#pragma unroll
        for (int i = 0; i < 8; ++i) t += red[i];
        c2[k] = t;
    }
}

// ---------------------------------------------------------------------------
// Kernel 2: fused GEMM + argmin.
//   Wave handles 16 rows of X. A fragments (16x256 bf16) cached in VGPRs,
//   converted from fp32 in flight. Loop over 64 iterations x two 16-center
//   tiles; each tile = 8x v_wmma_f32_16x16x32_bf16 (K=256 reduction).
//   Two independent accumulator chains break the WMMA RAW dependency so
//   B-tile loads overlap with matrix ops inside a single wave.
//   Candidate per (lane, accV): c2[col] - 2*dot  (||x||^2 is argmin-invariant).
//
//   A layout (16-bit A 16x32): lane<16 : K = t*32+0..15
//                              lane>=16: K = t*32+16..31
//   B layout (column per lane): lane%16 = column, half-wave selects K-half:
//                                          16 contiguous bf16 at K = t*32 + half*16
//   C/D layout: accV v, lanes 0-15 -> row v, lanes 16-31 -> row v+8; col = lane%16.
// ---------------------------------------------------------------------------
__global__ __launch_bounds__(256) void dcn_labels_kernel(
    const float*  __restrict__ X,
    const __bf16* __restrict__ Cb,
    const float*  __restrict__ c2,
    int*          __restrict__ labels_i,
    float*        __restrict__ labels_f) {
    const int lane = threadIdx.x & 31;
    const int wave = threadIdx.x >> 5;
    const int half = lane >> 4;          // 0 or 1
    const int l16  = lane & 15;
    const int rowbase = blockIdx.x * 128 + wave * 16;
    const int row = rowbase + l16;

    // ---- load + convert A fragments (row-major fp32 -> bf16 WMMA layout) ----
    v16bf a[8];
#pragma unroll
    for (int t = 0; t < 8; ++t) {
        const float4* p0 = (const float4*)(X + (size_t)row * DD + t * 32 + half * 16);
        const float4* p1 = (const float4*)(X + (size_t)row * DD + t * 32 + half * 16 + 8);
        float4 f0 = p0[0], f1 = p0[1], f2 = p1[0], f3 = p1[1];
        v16bf frag;
        frag[0]  = (__bf16)f0.x; frag[1]  = (__bf16)f0.y; frag[2]  = (__bf16)f0.z; frag[3]  = (__bf16)f0.w;
        frag[4]  = (__bf16)f1.x; frag[5]  = (__bf16)f1.y; frag[6]  = (__bf16)f1.z; frag[7]  = (__bf16)f1.w;
        frag[8]  = (__bf16)f2.x; frag[9]  = (__bf16)f2.y; frag[10] = (__bf16)f2.z; frag[11] = (__bf16)f2.w;
        frag[12] = (__bf16)f3.x; frag[13] = (__bf16)f3.y; frag[14] = (__bf16)f3.z; frag[15] = (__bf16)f3.w;
        a[t] = frag;
    }

    float best_d[8];
    int   best_i[8];
#pragma unroll
    for (int v = 0; v < 8; ++v) { best_d[v] = 3.4e38f; best_i[v] = 0; }

    // ---- sweep all K=2048 centers, two 16-wide tiles per iteration ----
    for (int ct = 0; ct < 64; ++ct) {
        const int col0 = ct * 32 + l16;
        const int col1 = col0 + 16;
        const __bf16* brow0 = Cb + (size_t)col0 * DD + half * 16;
        const __bf16* brow1 = Cb + (size_t)col1 * DD + half * 16;
        const float c2v0 = c2[col0];
        const float c2v1 = c2[col1];
        v8f acc0 = {};
        v8f acc1 = {};
#pragma unroll
        for (int t = 0; t < 8; ++t) {
            v16bf b0 = *(const v16bf*)(brow0 + t * 32);
            v16bf b1 = *(const v16bf*)(brow1 + t * 32);
            acc0 = __builtin_amdgcn_wmma_f32_16x16x32_bf16(
                false, a[t], false, b0, (short)0, acc0, false, false);
            acc1 = __builtin_amdgcn_wmma_f32_16x16x32_bf16(
                false, a[t], false, b1, (short)0, acc1, false, false);
        }
#pragma unroll
        for (int v = 0; v < 8; ++v) {
            float cand0 = __builtin_fmaf(-2.0f, acc0[v], c2v0);
            float cand1 = __builtin_fmaf(-2.0f, acc1[v], c2v1);
            if (cand0 < best_d[v]) { best_d[v] = cand0; best_i[v] = col0; }
            if (cand1 < best_d[v]) { best_d[v] = cand1; best_i[v] = col1; }
        }
    }

    // ---- min-argmin across the 16 lanes of each half (rows v / v+8) ----
#pragma unroll
    for (int v = 0; v < 8; ++v) {
        float d = best_d[v];
        int   i = best_i[v];
#pragma unroll
        for (int m = 1; m < 16; m <<= 1) {
            float od = __shfl_xor(d, m, 32);
            int   oi = __shfl_xor(i, m, 32);
            if (od < d || (od == d && oi < i)) { d = od; i = oi; }
        }
        best_d[v] = d; best_i[v] = i;
    }
    if (l16 == 0) {
#pragma unroll
        for (int v = 0; v < 8; ++v) {
            const int r = rowbase + half * 8 + v;
            labels_i[r] = best_i[v];
            labels_f[r] = (float)best_i[v];
        }
    }
}

// ---------------------------------------------------------------------------
// Kernel 3: segment sums + exact fp32 loss. One wave per row; lane owns 8 dims.
// ---------------------------------------------------------------------------
__global__ __launch_bounds__(256) void dcn_segment_kernel(
    const float* __restrict__ X, const float* __restrict__ C,
    const int* __restrict__ labels, float* __restrict__ seg,
    int* __restrict__ nass, float* __restrict__ loss) {
    const int lane = threadIdx.x & 31;
    const int nwaves = (gridDim.x * blockDim.x) >> 5;
    int wid = (blockIdx.x * blockDim.x + threadIdx.x) >> 5;
    for (int row = wid; row < NN; row += nwaves) {
        const int lab = labels[row];
        const float* xr = X + (size_t)row * DD + lane * 8;
        const float* cr = C + (size_t)lab * DD + lane * 8;
        float*       sr = seg + (size_t)lab * DD + lane * 8;
        float ls = 0.f;
#pragma unroll
        for (int e = 0; e < 8; ++e) {
            float x = xr[e];
            float df = x - cr[e];
            ls += df * df;
            unsafeAtomicAdd(&sr[e], x);
        }
#pragma unroll
        for (int m = 1; m < 32; m <<= 1) ls += __shfl_xor(ls, m, 32);
        if (lane == 0) {
            unsafeAtomicAdd(loss, ls);
            atomicAdd(&nass[lab], 1);
        }
    }
}

// ---------------------------------------------------------------------------
// Kernel 4: new_centers = (c0*m0 + seg)/(c0+n); new_counts; loss/N.
// ---------------------------------------------------------------------------
__global__ __launch_bounds__(256) void dcn_finalize_kernel(
    const float* __restrict__ C, const int* __restrict__ counts,
    const float* __restrict__ seg, const int* __restrict__ nass,
    const float* __restrict__ loss, float* __restrict__ out_loss,
    float* __restrict__ out_centers, float* __restrict__ out_counts) {
    const int k = blockIdx.x;
    const int d = threadIdx.x;
    const int c0 = counts[k];
    const int nc = c0 + nass[k];
    const float inv = 1.0f / (float)nc;
    out_centers[(size_t)k * DD + d] =
        ((float)c0 * C[(size_t)k * DD + d] + seg[(size_t)k * DD + d]) * inv;
    if (d == 0) out_counts[k] = (float)nc;
    if (k == 0 && d == 0) out_loss[0] = loss[0] / (float)NN;
}

// ---------------------------------------------------------------------------
extern "C" void kernel_launch(void* const* d_in, const int* in_sizes, int n_in,
                              void* d_out, int out_size, void* d_ws, size_t ws_size,
                              hipStream_t stream) {
    const float* X      = (const float*)d_in[0];   // [N, D] fp32
    const float* C      = (const float*)d_in[1];   // [K, D] fp32
    const int*   counts = (const int*)d_in[2];     // [K]    int32

    // d_out (float): labels[N], loss[1], new_centers[K*D], new_counts[K]
    float* out          = (float*)d_out;
    float* out_labels   = out;
    float* out_loss     = out + NN;
    float* out_centers  = out + NN + 1;
    float* out_counts   = out + NN + 1 + (size_t)KK * DD;

    // workspace layout (bytes)
    char* ws = (char*)d_ws;
    const size_t oCb   = 0;                               // K*D*2 = 1 MB
    const size_t oC2   = oCb   + (size_t)KK * DD * 2;     // K*4
    const size_t oLab  = oC2   + (size_t)KK * 4;          // N*4
    const size_t oSeg  = oLab  + (size_t)NN * 4;          // K*D*4
    const size_t oNass = oSeg  + (size_t)KK * DD * 4;     // K*4
    const size_t oLoss = oNass + (size_t)KK * 4;          // 4
    __bf16* Cb     = (__bf16*)(ws + oCb);
    float*  c2     = (float*) (ws + oC2);
    int*    labels = (int*)   (ws + oLab);
    float*  seg    = (float*) (ws + oSeg);
    int*    nass   = (int*)   (ws + oNass);
    float*  loss   = (float*) (ws + oLoss);

    // 0) zero atomic scratch (seg + nass + loss, contiguous)
    const int zwords = KK * DD + KK + 1;
    dcn_zero_kernel<<<1024, 256, 0, stream>>>(seg, zwords);

    // 1) centers -> bf16, c2
    dcn_prep_centers_kernel<<<KK, 256, 0, stream>>>(C, Cb, c2);

    // 2) fused WMMA GEMM + argmin -> labels
    dcn_labels_kernel<<<NN / 128, 256, 0, stream>>>(X, Cb, c2, labels, out_labels);

    // 3) segment sums + exact loss
    dcn_segment_kernel<<<1024, 256, 0, stream>>>(X, C, labels, seg, nass, loss);

    // 4) finalize centers / counts / loss
    dcn_finalize_kernel<<<KK, 256, 0, stream>>>(C, counts, seg, nass, loss,
                                                out_loss, out_centers, out_counts);
}
